// SPHADGCNN_67396626809177
// MI455X (gfx1250) — matrix operand order repaired and checked by reading, not compiled, not run on hardware
//
#include <hip/hip_runtime.h>

// ---------------- CDNA5 WMMA types ----------------
typedef __attribute__((ext_vector_type(16))) _Float16 v16h;
typedef __attribute__((ext_vector_type(8)))  _Float16 v8h;
typedef __attribute__((ext_vector_type(8)))  float    v8f;

#define Bn   4
#define Nn   4096
#define KNN  20
#define NKn  (Nn * KNN)
#define EPSb 1e-5f
#define SLP  0.2f

// ============================================================
// kNN: pd = -||xi - xj||^2, keep top-20 (stable on ties).
// One thread per query; 64-query x 64-candidate LDS tiles.
// grid (N/64, 1, B), block 64.
// ============================================================
__global__ void __launch_bounds__(64)
knn_kernel(const float* __restrict__ x, int* __restrict__ out, int C, int N) {
  __shared__ float sQ[64 * 64];
  __shared__ float sJ[64 * 64];
  const int tid = threadIdx.x;
  const int b   = blockIdx.z;
  const int i0  = blockIdx.x * 64;
  const float* xb = x + (size_t)b * C * N;

  for (int t = tid; t < C * 64; t += 64) {
    int c = t >> 6, q = t & 63;
    sQ[t] = xb[(size_t)c * N + (i0 + q)];
  }

  float bestv[KNN];
  int   besti[KNN];
#pragma unroll
  for (int s = 0; s < KNN; ++s) { bestv[s] = -3.4e38f; besti[s] = 0; }

  for (int j0 = 0; j0 < N; j0 += 64) {
    __syncthreads();
    for (int t = tid; t < C * 64; t += 64) {
      int c = t >> 6, j = t & 63;
      sJ[t] = xb[(size_t)c * N + (j0 + j)];
    }
    __syncthreads();
    for (int j = 0; j < 64; ++j) {
      float d = 0.0f;
      for (int c = 0; c < C; ++c) {
        float df = sQ[c * 64 + tid] - sJ[c * 64 + j];
        d = fmaf(df, df, d);
      }
      float pd = -d;
      if (pd > bestv[KNN - 1]) {
        int pos = KNN - 1;
        while (pos > 0 && bestv[pos - 1] < pd) {
          bestv[pos] = bestv[pos - 1]; besti[pos] = besti[pos - 1]; --pos;
        }
        bestv[pos] = pd; besti[pos] = j0 + j;
      }
    }
  }
  const int i = i0 + tid;
  for (int s = 0; s < KNN; ++s)
    out[((size_t)b * N + i) * KNN + s] = besti[s];
}

// ============================================================
// Edge-feature gather: out[b][0:C][n][k]=x[j]-x[n], [C:2C]=x[n],
// channels [2C:CP) zero-filled.  out has CP channels (f16).
// ============================================================
__global__ void gather_edge_kernel(const float* __restrict__ x,
                                   const int* __restrict__ idx,
                                   _Float16* __restrict__ out,
                                   int C, int CP, int N) {
  int t = blockIdx.x * blockDim.x + threadIdx.x;
  if (t >= N * KNN) return;
  const int b = blockIdx.z;
  const int n = t / KNN, kk = t % KNN;
  const int j = idx[((size_t)b * N + n) * KNN + kk];
  const float*  xb = x   + (size_t)b * C * N;
  _Float16*     ob = out + (size_t)b * CP * N * KNN;
  const size_t  pix = (size_t)n * KNN + kk;
  for (int c = 0; c < C; ++c) {
    float xn = xb[(size_t)c * N + n];
    float xj = xb[(size_t)c * N + j];
    ob[(size_t)c       * N * KNN + pix] = (_Float16)(xj - xn);
    ob[(size_t)(C + c) * N * KNN + pix] = (_Float16)xn;
  }
  for (int c = 2 * C; c < CP; ++c)
    ob[(size_t)c * N * KNN + pix] = (_Float16)0.0f;
}

// ============================================================
// car2sph on (x[j]-x[n]) of x_loc: ch0..5 = sph ++ (sph-mean),
// ch6..CP zero.  out [B,CP,N,K] f16.
// ============================================================
__global__ void car2sph_kernel(const float* __restrict__ x,
                               const int* __restrict__ idx,
                               _Float16* __restrict__ out, int CP, int N) {
  int t = blockIdx.x * blockDim.x + threadIdx.x;
  if (t >= N * KNN) return;
  const int b = blockIdx.z;
  const int n = t / KNN, kk = t % KNN;
  const int j = idx[((size_t)b * N + n) * KNN + kk];
  const float* xb = x + (size_t)b * 3 * N;
  float p0 = xb[j]         - xb[n];
  float p1 = xb[N + j]     - xb[N + n];
  float p2 = xb[2 * N + j] - xb[2 * N + n];
  float xy2   = p0 * p0 + p1 * p1;
  float r     = sqrtf(xy2 + p2 * p2);
  float theta = atan2f(p2, sqrtf(xy2));
  float phi   = atan2f(p1, p0);
  float mean  = (r + theta + phi) * (1.0f / 3.0f);
  _Float16* ob = out + (size_t)b * CP * N * KNN;
  const size_t pix = (size_t)n * KNN + kk;
  const size_t st  = (size_t)N * KNN;
  ob[0 * st + pix] = (_Float16)r;
  ob[1 * st + pix] = (_Float16)theta;
  ob[2 * st + pix] = (_Float16)phi;
  ob[3 * st + pix] = (_Float16)(r - mean);
  ob[4 * st + pix] = (_Float16)(theta - mean);
  ob[5 * st + pix] = (_Float16)(phi - mean);
  for (int c = 6; c < CP; ++c) ob[(size_t)c * st + pix] = (_Float16)0.0f;
}

// ============================================================
// Weight pack: f32 [O,Cin] -> f16 [Opad,Cpad], zero padded.
// ============================================================
__global__ void pack_w_kernel(const float* __restrict__ W, _Float16* __restrict__ Wp,
                              int O, int Cin, int Cpad, int total) {
  int t = blockIdx.x * blockDim.x + threadIdx.x;
  if (t >= total) return;
  int o = t / Cpad, c = t % Cpad;
  float v = (o < O && c < Cin) ? W[(size_t)o * Cin + c] : 0.0f;
  Wp[t] = (_Float16)v;
}

// BN fold: mode 1 -> (g*rsqrt(v+eps), b - m*scale); mode 2 -> (1, bias); mode 0 -> (1,0).
// Rows >= O get (0,0) so padded output rows are exactly zero.
__global__ void pack_bn_kernel(const float* __restrict__ g, const float* __restrict__ b,
                               const float* __restrict__ m, const float* __restrict__ v,
                               float2* __restrict__ SS, int O, int Opad, int mode) {
  int o = blockIdx.x * blockDim.x + threadIdx.x;
  if (o >= Opad) return;
  float sc = 0.0f, sh = 0.0f;
  if (o < O) {
    if (mode == 1) { sc = g[o] * rsqrtf(v[o] + EPSb); sh = b[o] - m[o] * sc; }
    else if (mode == 2) { sc = 1.0f; sh = b[o]; }
    else { sc = 1.0f; sh = 0.0f; }
  }
  SS[o] = make_float2(sc, sh);
}

// ============================================================
// WMMA GEMM: Y[b][yRows][M] = act( Wp[Opad,Cpad] @ X[b][Cpad,M] )
// Guard-free K-loop (padded weights/activations, M mult of 16).
// B operand staged via LDS with coalesced 16B loads shared by the
// whole block.  Each wave computes a 32(O) x 16(M) output tile:
// two WMMAs per B fragment (2x O-register-blocking) to halve the
// per-flop LDS/global B traffic.
// grid(ceil(M/64), ceil(yRows/32), batches), block 128.
// ============================================================
__global__ void __launch_bounds__(128)
wmma_gemm_kernel(const _Float16* __restrict__ X, const _Float16* __restrict__ Wp,
                 const float2* __restrict__ SS, void* __restrict__ Y,
                 int Cpad, int M, int yRows, int lrelu, int out_f32) {
  __shared__ _Float16 sX[32 * 64];
  const int tid  = threadIdx.x;
  const int lane = tid & 31;
  const int wave = tid >> 5;
  const int b    = blockIdx.z;
  const int mblk = blockIdx.x * 64;
  const int o0   = blockIdx.y * 32;
  const int half16 = lane >> 4;
  const int l16    = lane & 15;
  const int mcol   = wave * 16 + l16;

  const _Float16* Xb  = X  + (size_t)b * Cpad * M;
  const _Float16* Wr0 = Wp + (size_t)(o0 + l16) * Cpad;        // subtile 0 rows
  const _Float16* Wr1 = Wp + (size_t)(o0 + 16 + l16) * Cpad;   // subtile 1 rows

  // cooperative loader: 2 chunks of 8 halves per thread (32 rows x 64 cols)
  const int ch0 = tid * 2, ch1 = tid * 2 + 1;
  const int r0 = ch0 >> 3, c0 = (ch0 & 7) * 8;
  const int r1 = ch1 >> 3, c1 = (ch1 & 7) * 8;
  int mb0 = mblk + c0; if (mb0 > M - 8) mb0 = M - 8;
  int mb1 = mblk + c1; if (mb1 > M - 8) mb1 = M - 8;

  v8f acc0 = {}, acc1 = {};
  for (int kk = 0; kk < Cpad; kk += 32) {
    __syncthreads();
    *(v8h*)(&sX[r0 * 64 + c0]) = *(const v8h*)(Xb + (size_t)(kk + r0) * M + mb0);
    *(v8h*)(&sX[r1 * 64 + c1]) = *(const v8h*)(Xb + (size_t)(kk + r1) * M + mb1);
    if (kk + 32 < Cpad)
      __builtin_prefetch(Xb + (size_t)(kk + 32 + r0) * M + mb0, 0, 1);
    __syncthreads();

    // A fragments: aligned 16B loads (weights packed row-major, padded)
    v8h a0lo = *(const v8h*)(Wr0 + kk + half16 * 8);
    v8h a0hi = *(const v8h*)(Wr0 + kk + 16 + half16 * 8);
    v8h a1lo = *(const v8h*)(Wr1 + kk + half16 * 8);
    v8h a1hi = *(const v8h*)(Wr1 + kk + 16 + half16 * 8);
    v16h a0, a1, bf;
#pragma unroll
    for (int h = 0; h < 8; ++h) {
      a0[h] = a0lo[h]; a0[h + 8] = a0hi[h];
      a1[h] = a1lo[h]; a1[h + 8] = a1hi[h];
    }
    // B fragment: lane = column, K in table order matching A (from LDS)
#pragma unroll
    for (int h = 0; h < 16; ++h) bf[h] = sX[((h < 8) ? (half16 * 8 + h) : (16 + half16 * 8 + h - 8)) * 64 + mcol];

    acc0 = __builtin_amdgcn_wmma_f32_16x16x32_f16(false, a0, false, bf,
                                                  (short)0, acc0, false, false);
    acc1 = __builtin_amdgcn_wmma_f32_16x16x32_f16(false, a1, false, bf,
                                                  (short)0, acc1, false, false);
  }

  const int m = mblk + mcol;
#pragma unroll
  for (int r = 0; r < 8; ++r) {
#pragma unroll
    for (int sub = 0; sub < 2; ++sub) {
      int o = o0 + sub * 16 + half16 * 8 + r;
      float av = (sub == 0) ? acc0[r] : acc1[r];
      if (o < yRows && m < M) {
        float2 ss = SS[o];
        float v = fmaf(av, ss.x, ss.y);
        if (lrelu) v = (v >= 0.0f) ? v : SLP * v;
        size_t offo = (size_t)b * yRows * M + (size_t)o * M + m;
        if (out_f32) ((float*)Y)[offo] = v;
        else         ((_Float16*)Y)[offo] = (_Float16)v;
      }
    }
  }
}

// ============================================================
// max over k: [B,C,N,K] f16 -> [B,C,N] f16
// ============================================================
__global__ void max_k_kernel(const _Float16* __restrict__ in,
                             _Float16* __restrict__ out, int C, int N) {
  int t = blockIdx.x * blockDim.x + threadIdx.x;
  if (t >= C * N) return;
  const int b = blockIdx.z;
  const _Float16* p = in + ((size_t)b * C * N + t) * KNN;
  float m = -3.4e38f;
  for (int k = 0; k < KNN; ++k) m = fmaxf(m, (float)p[k]);
  out[(size_t)b * C * N + t] = (_Float16)m;
}

// ============================================================
// max over N: [B,C,N] f16 -> [C x MS] f16 at column b (wave per (b,c))
// ============================================================
__global__ void __launch_bounds__(32)
max_n_kernel(const _Float16* __restrict__ in, _Float16* __restrict__ out,
             int C, int N, int MS) {
  const int c = blockIdx.x, b = blockIdx.z, lane = threadIdx.x;
  const _Float16* p = in + ((size_t)b * C + c) * N;
  float m = -3.4e38f;
  for (int n = lane; n < N; n += 32) m = fmaxf(m, (float)p[n]);
  for (int s = 16; s; s >>= 1) m = fmaxf(m, __shfl_xor(m, s, 32));
  if (lane == 0) out[(size_t)c * MS + b] = (_Float16)m;
}

// ============================================================
// attention pool finish: out[b,c,n] = sum_k softmax_k(h) * h
// ============================================================
__global__ void attpool_kernel(const _Float16* __restrict__ h,
                               float* __restrict__ out, int C, int N) {
  int t = blockIdx.x * blockDim.x + threadIdx.x;
  if (t >= C * N) return;
  const int b = blockIdx.z;
  const _Float16* p = h + ((size_t)b * C * N + t) * KNN;
  float mx = -3.4e38f;
  for (int k = 0; k < KNN; ++k) mx = fmaxf(mx, (float)p[k]);
  float se = 0.0f, acc = 0.0f;
  for (int k = 0; k < KNN; ++k) {
    float hv = (float)p[k];
    float e  = expf(hv - mx);
    se += e; acc += e * hv;
  }
  out[(size_t)b * C * N + t] = acc / se;
}

// ============================================================
// sphprop finish: out = x + sum_k x[:, idx[n,k]] * softmax_k(logit)
// logit buffer has 16 channels, channel 0 used.
// ============================================================
__global__ void sphprop_kernel(const float* __restrict__ xin,
                               const _Float16* __restrict__ logit,
                               const int* __restrict__ idx,
                               float* __restrict__ out, int C, int N) {
  int t = blockIdx.x * blockDim.x + threadIdx.x;
  if (t >= C * N) return;
  const int b = blockIdx.z;
  const int c = t / N, n = t % N;
  const _Float16* lp = logit + (size_t)b * 16 * (size_t)N * KNN + (size_t)n * KNN;
  float mx = -3.4e38f;
  for (int k = 0; k < KNN; ++k) mx = fmaxf(mx, (float)lp[k]);
  float se = 0.0f;
  for (int k = 0; k < KNN; ++k) se += expf((float)lp[k] - mx);
  const int*   ip = idx + ((size_t)b * N + n) * KNN;
  const float* xb = xin + ((size_t)b * C + c) * N;
  float acc = 0.0f;
  for (int k = 0; k < KNN; ++k)
    acc += expf((float)lp[k] - mx) * xb[ip[k]];
  out[((size_t)b * C + c) * N + n] = xb[n] + acc / se;
}

// ============================================================
// apply 3x3 transform: xt[b,d,n] = sum_c x[b,c,n] * T[(3c+d)*16 + b]
// T is the ttr GEMM output laid out [32 rows, 16 cols].
// ============================================================
__global__ void transform_kernel(const float* __restrict__ x,
                                 const float* __restrict__ T,
                                 float* __restrict__ out, int N) {
  int n = blockIdx.x * blockDim.x + threadIdx.x;
  if (n >= N) return;
  const int b = blockIdx.z;
  const float* xb = x + (size_t)b * 3 * N;
  float x0 = xb[n], x1 = xb[N + n], x2 = xb[2 * N + n];
  for (int d = 0; d < 3; ++d) {
    float v = x0 * T[(0 + d) * 16 + b] + x1 * T[(3 + d) * 16 + b] + x2 * T[(6 + d) * 16 + b];
    out[(size_t)b * 3 * N + (size_t)d * N + n] = v;
  }
}

// concat [x1,x2,x3] (each [B,64,N] f32) -> [B,192,N] f16
__global__ void concat3_kernel(const float* __restrict__ x1, const float* __restrict__ x2,
                               const float* __restrict__ x3, _Float16* __restrict__ out,
                               int N) {
  int t = blockIdx.x * blockDim.x + threadIdx.x;
  if (t >= 192 * N) return;
  const int b = blockIdx.z;
  const int c = t / N, n = t % N;
  const float* s = (c < 64) ? x1 : ((c < 128) ? x2 : x3);
  out[(size_t)b * 192 * N + t] = (_Float16)s[((size_t)b * 64 + (c & 63)) * N + n];
}

// concat [g(bcast), x1,x2,x3] -> [B,1216,N] f16 ; g is [1024 x 16] f16, col b
__global__ void concat_g_kernel(const _Float16* __restrict__ g,
                                const float* __restrict__ x1, const float* __restrict__ x2,
                                const float* __restrict__ x3, _Float16* __restrict__ out,
                                int N) {
  int t = blockIdx.x * blockDim.x + threadIdx.x;
  if (t >= 1216 * N) return;
  const int b = blockIdx.z;
  const int c = t / N, n = t % N;
  _Float16 v;
  if (c < 1024) v = g[(size_t)c * 16 + b];
  else {
    int cc = c - 1024;
    const float* s = (cc < 64) ? x1 : ((cc < 128) ? x2 : x3);
    v = (_Float16)s[((size_t)b * 64 + (cc & 63)) * N + n];
  }
  out[(size_t)b * 1216 * N + t] = v;
}

// ---------------- host-side plumbing ----------------
struct BNp { const float *g, *b, *m, *v; };
struct Packed { _Float16* Wp; float2* SS; int Cpad, Opad; };

static Packed prep(hipStream_t s, char* wsb, size_t& off, const float* W, int O, int Cin,
                   const float* g, const float* b, const float* m, const float* v,
                   int mode, int yRows) {
  Packed p;
  p.Cpad = (Cin + 31) & ~31;
  p.Opad = (yRows + 31) & ~31;       // 32-row O-blocking
  p.Wp = (_Float16*)(wsb + off);
  off = (off + (size_t)p.Opad * p.Cpad * 2 + 255) & ~(size_t)255;
  p.SS = (float2*)(wsb + off);
  off = (off + (size_t)p.Opad * sizeof(float2) + 255) & ~(size_t)255;
  int tot = p.Opad * p.Cpad;
  pack_w_kernel<<<(tot + 255) / 256, 256, 0, s>>>(W, p.Wp, O, Cin, p.Cpad, tot);
  pack_bn_kernel<<<(p.Opad + 255) / 256, 256, 0, s>>>(g, b, m, v, p.SS, O, p.Opad, mode);
  return p;
}

static void rungemm(hipStream_t s, const _Float16* X, const Packed& p, void* Y,
                    int M, int batches, int yRows, int lrelu, int out_f32) {
  dim3 grid((M + 63) / 64, (yRows + 31) / 32, batches);
  wmma_gemm_kernel<<<grid, 128, 0, s>>>(X, p.Wp, p.SS, Y, p.Cpad, M, yRows, lrelu, out_f32);
}

extern "C" void kernel_launch(void* const* d_in, const int* in_sizes, int n_in,
                              void* d_out, int out_size, void* d_ws, size_t ws_size,
                              hipStream_t stream) {
  (void)in_sizes; (void)n_in; (void)out_size; (void)ws_size;
  const int N = Nn, B = Bn;

  // ---- unpack params in setup_inputs() insertion order ----
  int ci = 0;
  auto nxt = [&]() { return (const float*)d_in[ci++]; };
  auto nbn = [&]() { BNp r; r.g = nxt(); r.b = nxt(); r.m = nxt(); r.v = nxt(); return r; };

  const float* x_input = nxt();
  const float* tc1_w = nxt(); BNp tc1 = nbn();
  const float* tc2_w = nxt(); BNp tc2 = nbn();
  const float* tc3_w = nxt(); BNp tc3 = nbn();
  const float* tl1_w = nxt(); BNp tl1 = nbn();
  const float* tl2_w = nxt(); BNp tl2 = nbn();
  const float* ttr_w = nxt(); const float* ttr_b = nxt();
  const float* c1_w  = nxt(); BNp c1  = nbn();
  const float* c2_w  = nxt(); BNp c2  = nbn();
  const float* c3_w  = nxt(); BNp c3  = nbn();
  const float* c4_w  = nxt(); BNp c4  = nbn();
  const float* c5_w  = nxt(); BNp c5  = nbn();
  const float* c6_w  = nxt(); BNp c6  = nbn();
  const float* c8_w  = nxt(); BNp c8  = nbn();
  const float* c9_w  = nxt(); BNp c9  = nbn();
  const float* c10_w = nxt(); BNp c10 = nbn();
  const float* c11_w = nxt();
  struct APp { const float* w1; BNp b1; const float* w2; BNp b2; } ap[3];
  for (int i = 0; i < 3; ++i) { ap[i].w1 = nxt(); ap[i].b1 = nbn(); ap[i].w2 = nxt(); ap[i].b2 = nbn(); }
  struct SPp { const float* w1; BNp b1; const float* w2; BNp b2; const float* w3; BNp b3; } sp[3];
  for (int i = 0; i < 3; ++i) { sp[i].w1 = nxt(); sp[i].b1 = nbn(); sp[i].w2 = nxt(); sp[i].b2 = nbn();
                                sp[i].w3 = nxt(); sp[i].b3 = nbn(); }

  // ---- workspace carve ----
  char* wsb = (char*)d_ws;
  size_t off = 0;
  auto alloc = [&](size_t bytes) { void* p = wsb + off; off = (off + bytes + 255) & ~(size_t)255; return p; };

  // packed weights + folded BN (f16 / float2; ~3 MB total)
  Packed Ptc1 = prep(stream, wsb, off, tc1_w, 64, 6,    tc1.g, tc1.b, tc1.m, tc1.v, 1, 64);
  Packed Ptc2 = prep(stream, wsb, off, tc2_w, 128, 64,  tc2.g, tc2.b, tc2.m, tc2.v, 1, 128);
  Packed Ptc3 = prep(stream, wsb, off, tc3_w, 1024, 128,tc3.g, tc3.b, tc3.m, tc3.v, 1, 1024);
  Packed Ptl1 = prep(stream, wsb, off, tl1_w, 512, 1024,tl1.g, tl1.b, tl1.m, tl1.v, 1, 512);
  Packed Ptl2 = prep(stream, wsb, off, tl2_w, 256, 512, tl2.g, tl2.b, tl2.m, tl2.v, 1, 256);
  Packed Pttr = prep(stream, wsb, off, ttr_w, 9, 256,   nullptr, ttr_b, nullptr, nullptr, 2, 16);
  Packed Pc1  = prep(stream, wsb, off, c1_w, 64, 6,     c1.g, c1.b, c1.m, c1.v, 1, 64);
  Packed Pc2  = prep(stream, wsb, off, c2_w, 64, 64,    c2.g, c2.b, c2.m, c2.v, 1, 64);
  Packed Pc3  = prep(stream, wsb, off, c3_w, 64, 128,   c3.g, c3.b, c3.m, c3.v, 1, 64);
  Packed Pc4  = prep(stream, wsb, off, c4_w, 64, 64,    c4.g, c4.b, c4.m, c4.v, 1, 64);
  Packed Pc5  = prep(stream, wsb, off, c5_w, 64, 128,   c5.g, c5.b, c5.m, c5.v, 1, 64);
  Packed Pc6  = prep(stream, wsb, off, c6_w, 1024, 192, c6.g, c6.b, c6.m, c6.v, 1, 1024);
  Packed Pc8  = prep(stream, wsb, off, c8_w, 256, 1216, c8.g, c8.b, c8.m, c8.v, 1, 256);
  Packed Pc9  = prep(stream, wsb, off, c9_w, 256, 256,  c9.g, c9.b, c9.m, c9.v, 1, 256);
  Packed Pc10 = prep(stream, wsb, off, c10_w, 128, 256, c10.g, c10.b, c10.m, c10.v, 1, 128);
  Packed Pc11 = prep(stream, wsb, off, c11_w, 5, 128,   nullptr, nullptr, nullptr, nullptr, 0, 5);
  Packed Pap1[3], Pap2[3];
  for (int i = 0; i < 3; ++i) {
    Pap1[i] = prep(stream, wsb, off, ap[i].w1, 32, 64, ap[i].b1.g, ap[i].b1.b, ap[i].b1.m, ap[i].b1.v, 1, 32);
    Pap2[i] = prep(stream, wsb, off, ap[i].w2, 64, 32, ap[i].b2.g, ap[i].b2.b, ap[i].b2.m, ap[i].b2.v, 1, 64);
  }
  Packed Psp1[3], Psp2[3], Psp3[3];
  for (int i = 0; i < 3; ++i) {
    Psp1[i] = prep(stream, wsb, off, sp[i].w1, 6, 6, sp[i].b1.g, sp[i].b1.b, sp[i].b1.m, sp[i].b1.v, 1, 32);
    Psp2[i] = prep(stream, wsb, off, sp[i].w2, 3, 6, sp[i].b2.g, sp[i].b2.b, sp[i].b2.m, sp[i].b2.v, 1, 32);
    Psp3[i] = prep(stream, wsb, off, sp[i].w3, 1, 3, sp[i].b3.g, sp[i].b3.b, sp[i].b3.m, sp[i].b3.v, 1, 16);
  }

  // activation buffers (channel counts padded to consumer Cpad)
  int*      IDX0  = (int*)alloc((size_t)B * N * KNN * 4);
  int*      IDX1  = (int*)alloc((size_t)B * N * KNN * 4);
  int*      IDX2  = (int*)alloc((size_t)B * N * KNN * 4);
  int*      IDX3  = (int*)alloc((size_t)B * N * KNN * 4);
  _Float16* GEO   = (_Float16*)alloc((size_t)B * 32 * NKn * 2);
  _Float16* BUF6  = (_Float16*)alloc((size_t)B * 32 * NKn * 2);
  _Float16* B64A  = (_Float16*)alloc((size_t)B * 64 * NKn * 2);
  _Float16* B64B  = (_Float16*)alloc((size_t)B * 64 * NKn * 2);
  _Float16* B128  = (_Float16*)alloc((size_t)B * 128 * NKn * 2);
  _Float16* B32   = (_Float16*)alloc((size_t)B * 32 * NKn * 2);
  _Float16* S1    = (_Float16*)alloc((size_t)B * 32 * NKn * 2);
  _Float16* S2    = (_Float16*)alloc((size_t)B * 32 * NKn * 2);
  _Float16* S3    = (_Float16*)alloc((size_t)B * 16 * NKn * 2);
  _Float16* TM    = (_Float16*)alloc((size_t)B * 128 * N * 2);
  _Float16* T1024 = (_Float16*)alloc((size_t)B * 1024 * N * 2);
  _Float16* TG    = (_Float16*)alloc((size_t)1024 * 16 * 2);
  _Float16* L1    = (_Float16*)alloc((size_t)512 * 16 * 2);
  _Float16* L2    = (_Float16*)alloc((size_t)256 * 16 * 2);
  float*    TMAT  = (float*)alloc((size_t)32 * 16 * 4);
  float*    XT    = (float*)alloc((size_t)B * 3 * N * 4);
  float*    P     = (float*)alloc((size_t)B * 64 * N * 4);
  float*    X1    = (float*)alloc((size_t)B * 64 * N * 4);
  float*    X2    = (float*)alloc((size_t)B * 64 * N * 4);
  float*    X3    = (float*)alloc((size_t)B * 64 * N * 4);
  _Float16* X192  = (_Float16*)alloc((size_t)B * 192 * N * 2);
  _Float16* X1216 = (_Float16*)alloc((size_t)B * 1216 * N * 2);
  _Float16* C8o   = (_Float16*)alloc((size_t)B * 256 * N * 2);
  _Float16* C9o   = (_Float16*)alloc((size_t)B * 256 * N * 2);
  _Float16* C10o  = (_Float16*)alloc((size_t)B * 128 * N * 2);

  const dim3 gNK((NKn + 255) / 256, 1, B), bl(256);
  const dim3 gKNN(N / 64, 1, B);
  const dim3 gCN64(((64 * N) + 255) / 256, 1, B);

  // ---- stage 0: kNN on input + invariant spherical features ----
  knn_kernel<<<gKNN, 64, 0, stream>>>(x_input, IDX0, 3, N);
  car2sph_kernel<<<gNK, bl, 0, stream>>>(x_input, IDX0, GEO, 32, N);
  gather_edge_kernel<<<gNK, bl, 0, stream>>>(x_input, IDX0, BUF6, 3, 32, N);

  // ---- transform net ----
  rungemm(stream, BUF6, Ptc1, B64A, NKn, B, 64, 1, 0);
  rungemm(stream, B64A, Ptc2, B128, NKn, B, 128, 1, 0);
  { dim3 g(((128 * N) + 255) / 256, 1, B);
    max_k_kernel<<<g, bl, 0, stream>>>(B128, TM, 128, N); }
  rungemm(stream, TM, Ptc3, T1024, N, B, 1024, 1, 0);
  { dim3 g(1024, 1, B); max_n_kernel<<<g, 32, 0, stream>>>(T1024, TG, 1024, N, 16); }
  rungemm(stream, TG, Ptl1, L1, 16, 1, 512, 1, 0);
  rungemm(stream, L1, Ptl2, L2, 16, 1, 256, 1, 0);
  rungemm(stream, L2, Pttr, TMAT, 16, 1, 16, 0, 1);
  { dim3 g((N + 255) / 256, 1, B);
    transform_kernel<<<g, bl, 0, stream>>>(x_input, TMAT, XT, N); }

  // ---- block 1 ----
  knn_kernel<<<gKNN, 64, 0, stream>>>(XT, IDX1, 3, N);
  gather_edge_kernel<<<gNK, bl, 0, stream>>>(XT, IDX1, BUF6, 3, 32, N);
  rungemm(stream, BUF6, Pc1, B64A, NKn, B, 64, 1, 0);
  rungemm(stream, B64A, Pc2, B64B, NKn, B, 64, 1, 0);
  rungemm(stream, B64B, Pap1[0], B32, NKn, B, 32, 1, 0);
  rungemm(stream, B32,  Pap2[0], B64A, NKn, B, 64, 1, 0);
  attpool_kernel<<<gCN64, bl, 0, stream>>>(B64A, P, 64, N);
  rungemm(stream, GEO, Psp1[0], S1, NKn, B, 32, 1, 0);
  rungemm(stream, S1,  Psp2[0], S2, NKn, B, 32, 1, 0);
  rungemm(stream, S2,  Psp3[0], S3, NKn, B, 16, 1, 0);
  sphprop_kernel<<<gCN64, bl, 0, stream>>>(P, S3, IDX0, X1, 64, N);

  // ---- block 2 ----
  knn_kernel<<<gKNN, 64, 0, stream>>>(X1, IDX2, 64, N);
  gather_edge_kernel<<<gNK, bl, 0, stream>>>(X1, IDX2, B128, 64, 128, N);
  rungemm(stream, B128, Pc3, B64A, NKn, B, 64, 1, 0);
  rungemm(stream, B64A, Pc4, B64B, NKn, B, 64, 1, 0);
  rungemm(stream, B64B, Pap1[1], B32, NKn, B, 32, 1, 0);
  rungemm(stream, B32,  Pap2[1], B64A, NKn, B, 64, 1, 0);
  attpool_kernel<<<gCN64, bl, 0, stream>>>(B64A, P, 64, N);
  rungemm(stream, GEO, Psp1[1], S1, NKn, B, 32, 1, 0);
  rungemm(stream, S1,  Psp2[1], S2, NKn, B, 32, 1, 0);
  rungemm(stream, S2,  Psp3[1], S3, NKn, B, 16, 1, 0);
  sphprop_kernel<<<gCN64, bl, 0, stream>>>(P, S3, IDX0, X2, 64, N);

  // ---- block 3 ----
  knn_kernel<<<gKNN, 64, 0, stream>>>(X2, IDX3, 64, N);
  gather_edge_kernel<<<gNK, bl, 0, stream>>>(X2, IDX3, B128, 64, 128, N);
  rungemm(stream, B128, Pc5, B64B, NKn, B, 64, 1, 0);
  rungemm(stream, B64B, Pap1[2], B32, NKn, B, 32, 1, 0);
  rungemm(stream, B32,  Pap2[2], B64A, NKn, B, 64, 1, 0);
  attpool_kernel<<<gCN64, bl, 0, stream>>>(B64A, P, 64, N);
  rungemm(stream, GEO, Psp1[2], S1, NKn, B, 32, 1, 0);
  rungemm(stream, S1,  Psp2[2], S2, NKn, B, 32, 1, 0);
  rungemm(stream, S2,  Psp3[2], S3, NKn, B, 16, 1, 0);
  sphprop_kernel<<<gCN64, bl, 0, stream>>>(P, S3, IDX0, X3, 64, N);

  // ---- head ----
  { dim3 g(((192 * N) + 255) / 256, 1, B);
    concat3_kernel<<<g, bl, 0, stream>>>(X1, X2, X3, X192, N); }
  rungemm(stream, X192, Pc6, T1024, N, B, 1024, 1, 0);
  { dim3 g(1024, 1, B); max_n_kernel<<<g, 32, 0, stream>>>(T1024, TG, 1024, N, 16); }
  { dim3 g(((1216 * N) + 255) / 256, 1, B);
    concat_g_kernel<<<g, bl, 0, stream>>>(TG, X1, X2, X3, X1216, N); }
  rungemm(stream, X1216, Pc8,  C8o,  N, B, 256, 1, 0);
  rungemm(stream, C8o,   Pc9,  C9o,  N, B, 256, 1, 0);
  rungemm(stream, C9o,   Pc10, C10o, N, B, 128, 1, 0);
  rungemm(stream, C10o,  Pc11, d_out, N, B, 5, 0, 1);
}
